// AttnBlock_4406636445827
// MI455X (gfx1250) — compile-verified, matrix operand order checked by reading, not run on hardware
//
#include <hip/hip_runtime.h>

// ---------------------------------------------------------------------------
// AttnBlock for MI455X (gfx1250, wave32). All GEMMs use v_wmma_f32_16x16x32_f16.
// Workspace layout (~210 MB):
//   [0,1024)        : 128 group means + 128 group rstds (f32)
//   xn  f16 [B,N,C] : 16 MB   (GroupNorm output, pixel-major)
//   q   f16 [B,N,C] : 16 MB   (pre-scaled by C^-0.5)
//   k   f16 [B,N,C] : 16 MB
//   vT  f16 [B,C,N] : 16 MB   (channel-major -> B-fragments for P*V)
//   P   f16 [B,N,N] : 128 MB  (normalized attention)
//   O   f16 [B,N,C] : 16 MB
//   Wh  f16 4x[C,C] : 2 MB    (pre-converted wq/wk/wv/wo)
// ---------------------------------------------------------------------------

#define CC   512
#define NPIX 4096   // 64*64
#define BB   4
#define GRPS 32
#define QSCALE 0.04419417382415922f   // 512^-0.5

typedef __attribute__((ext_vector_type(16))) _Float16 v16h;
typedef __attribute__((ext_vector_type(8)))  _Float16 v8h;
typedef __attribute__((ext_vector_type(8)))  float    v8f;
typedef __attribute__((ext_vector_type(4)))  float    v4f;

// Load a 16x32 f16 A-fragment (or 32x16 B-fragment: rows of the stored matrix
// are the K-major vectors of one column). ISA layout (CDNA5 ISA):
//   lanes 0-15 : elems 0..7 -> K=kb..kb+7,   elems 8..15 -> K=kb+16..kb+23
//   lanes16-31 : elems 0..7 -> K=kb+8..+15,  elems 8..15 -> K=kb+24..+31
__device__ __forceinline__ v16h frag_f16(const _Float16* base, int ld, int row0,
                                         int kb, int lane) {
  const int lr = lane & 15, hf = lane >> 4;
  const _Float16* p = base + (size_t)(row0 + lr) * ld + kb + hf * 8;
  v8h lo = *(const v8h*)p;
  v8h hi = *(const v8h*)(p + 16);
  v16h r;
#pragma unroll
  for (int i = 0; i < 8; ++i) { r[i] = lo[i]; r[i + 8] = hi[i]; }
  return r;
}

#define WMMA_F16(a, b, c) \
  __builtin_amdgcn_wmma_f32_16x16x32_f16(false, (a), false, (b), (short)0, (c), false, false)

// ---------------- 0) Pre-convert all four weight matrices to f16 ------------
__global__ __launch_bounds__(256) void prep_weights(const float* __restrict__ wq,
                                                    const float* __restrict__ wk,
                                                    const float* __restrict__ wv,
                                                    const float* __restrict__ wo,
                                                    _Float16* __restrict__ wh) {
  const int i = (blockIdx.x * 256 + threadIdx.x) * 4;   // over 4*C*C elements
  const int which = i >> 18;                            // /(512*512)
  const int off = i & (CC * CC - 1);
  const float* src = (which == 0) ? wq : (which == 1) ? wk : (which == 2) ? wv : wo;
  v4f v = *(const v4f*)(src + off);
  _Float16* d = wh + (size_t)which * CC * CC + off;
#pragma unroll
  for (int t = 0; t < 4; ++t) d[t] = (_Float16)v[t];
}

// ---------------- 1) GroupNorm statistics: one block per (batch, group) -----
__global__ __launch_bounds__(256) void gn_stats(const float* __restrict__ x,
                                                float* __restrict__ stats) {
  __shared__ float s1[256], s2[256];
  const int bg = blockIdx.x;               // b*32+g
  const float* xp = x + (size_t)bg * 16 * NPIX;  // group = 16 contiguous channels
  float sum = 0.f, sq = 0.f;
  for (int i = threadIdx.x; i < 16 * NPIX; i += 256) {
    float v = xp[i];
    sum += v; sq += v * v;
  }
  s1[threadIdx.x] = sum; s2[threadIdx.x] = sq;
  __syncthreads();
  for (int off = 128; off > 0; off >>= 1) {
    if (threadIdx.x < off) {
      s1[threadIdx.x] += s1[threadIdx.x + off];
      s2[threadIdx.x] += s2[threadIdx.x + off];
    }
    __syncthreads();
  }
  if (threadIdx.x == 0) {
    const float inv = 1.0f / (16.0f * NPIX);
    float mean = s1[0] * inv;
    float var  = s2[0] * inv - mean * mean;
    stats[bg]       = mean;
    stats[128 + bg] = rsqrtf(var + 1e-6f);
  }
}

// ---------------- 2) GN apply + transpose to xn f16 [B,N,C] -----------------
__global__ __launch_bounds__(256) void gn_apply(const float* __restrict__ x,
                                                const float* __restrict__ gw,
                                                const float* __restrict__ gb,
                                                const float* __restrict__ stats,
                                                _Float16* __restrict__ xn) {
  const int bc = blockIdx.x;                // b*512 + c
  const int b = bc >> 9, c = bc & 511;
  const int bg = b * GRPS + (c >> 4);
  const float mean = stats[bg], rstd = stats[128 + bg];
  const float sc = rstd * gw[c];
  const float sh = gb[c] - mean * sc;
  const float* xp = x + (size_t)bc * NPIX;
  _Float16* xo = xn + (size_t)b * NPIX * CC + c;
  for (int p = threadIdx.x; p < NPIX; p += 256)
    xo[(size_t)p * CC] = (_Float16)(xp[p] * sc + sh);
}

// ---------------- 3) Q/K/V projections (wave: 16 pixels x 64 couts) ---------
// grid (256 m-tiles, 6, B); 4 waves/block -> strip s in [0,24): s/8 picks q/k/v.
__global__ __launch_bounds__(128) void qkv_gemm(const _Float16* __restrict__ xn,
                                                const _Float16* __restrict__ wh,
                                                const float* __restrict__ bq,
                                                const float* __restrict__ bk,
                                                const float* __restrict__ bv,
                                                _Float16* __restrict__ q,
                                                _Float16* __restrict__ k,
                                                _Float16* __restrict__ vT) {
  const int lane = threadIdx.x & 31, wave = threadIdx.x >> 5;
  const int s = blockIdx.y * 4 + wave;      // 0..23
  const int which = s >> 3;                 // 0=q 1=k 2=v
  const int nb = (s & 7) * 64;
  const int mb = blockIdx.x * 16;
  const int b  = blockIdx.z;
  const _Float16* W = wh + (size_t)which * CC * CC;
  const float* bs = (which == 0) ? bq : (which == 1) ? bk : bv;
  const _Float16* A = xn + (size_t)b * NPIX * CC;

  v8f acc[4] = {};
  for (int kb2 = 0; kb2 < CC; kb2 += 32) {
    v16h a = frag_f16(A, CC, mb, kb2, lane);
#pragma unroll
    for (int t = 0; t < 4; ++t) {
      v16h bf = frag_f16(W, CC, nb + t * 16, kb2, lane);
      acc[t] = WMMA_F16(a, bf, acc[t]);
    }
  }
  const int lr = lane & 15, hf = lane >> 4;
  _Float16* dq = (which == 0) ? q : k;
#pragma unroll
  for (int t = 0; t < 4; ++t) {
    const int n = nb + t * 16 + lr;
    const float bias = bs[n];
#pragma unroll
    for (int v = 0; v < 8; ++v) {
      const int m = mb + v + 8 * hf;
      float val = acc[t][v] + bias;
      if (which == 0) val *= QSCALE;                      // fold softmax scale
      if (which < 2)
        dq[((size_t)b * NPIX + m) * CC + n] = (_Float16)val;   // [N,C]
      else
        vT[((size_t)b * CC + n) * NPIX + m] = (_Float16)val;   // [C,N]
    }
  }
}

// ---------------- 4) Scores + row softmax: block owns 16 query rows ---------
// 8 waves x 512 keys each. Q panel (16x512 f16 = 128 VGPRs/lane) cached in
// registers. Two j-tiles processed concurrently so one tile's WMMAs overlap
// the other tile's K-fragment loads (avoids full s_wait_loadcnt 0 stalls).
// Score strip (16x4096 f32) lives in 256KB dynamic LDS (WGP has 320KB).
__global__ __launch_bounds__(256) void attn_scores_softmax(
    const _Float16* __restrict__ q, const _Float16* __restrict__ k,
    _Float16* __restrict__ P) {
  extern __shared__ float srow[];           // 16 * 4096 f32 = 256 KB
  __shared__ float red[16 * 16];
  __shared__ float rowm[16], rowinv[16];
  const int b = blockIdx.y, mb = blockIdx.x * 16;
  const int tid = threadIdx.x, wave = tid >> 5, lane = tid & 31;
  const int lr = lane & 15, hf = lane >> 4;
  const _Float16* Qb = q + (size_t)b * NPIX * CC;
  const _Float16* Kb = k + (size_t)b * NPIX * CC;

  v16h aq[16];                              // whole 16x512 Q panel, registers
#pragma unroll
  for (int t = 0; t < 16; ++t) aq[t] = frag_f16(Qb, CC, mb, t * 32, lane);

  for (int jt = 0; jt < 32; jt += 2) {      // two 16-col tiles in flight
    const int j0 = wave * 512 + jt * 16;
    // prefetch the next pair of key rows toward the WGP
    __builtin_prefetch(Kb + (size_t)(j0 + 32 + lr) * CC, 0, 1);
    __builtin_prefetch(Kb + (size_t)(j0 + 48 + lr) * CC, 0, 1);
    v8f acc0 = {}, acc1 = {};
#pragma unroll
    for (int t = 0; t < 16; ++t) {
      v16h bf0 = frag_f16(Kb, CC, j0,      t * 32, lane);
      v16h bf1 = frag_f16(Kb, CC, j0 + 16, t * 32, lane);
      acc0 = WMMA_F16(aq[t], bf0, acc0);
      acc1 = WMMA_F16(aq[t], bf1, acc1);
    }
#pragma unroll
    for (int v = 0; v < 8; ++v) {
      srow[(v + 8 * hf) * NPIX + j0 + lr]      = acc0[v];
      srow[(v + 8 * hf) * NPIX + j0 + 16 + lr] = acc1[v];
    }
  }
  __syncthreads();

  const int r = tid >> 4, sub = tid & 15;   // 16 threads per row
  const float* rp = srow + r * NPIX + sub * 256;
  float mx = -3.4e38f;
  for (int j = 0; j < 256; ++j) mx = fmaxf(mx, rp[j]);
  red[r * 16 + sub] = mx;
  __syncthreads();
  if (sub == 0) {
    float mm = red[r * 16];
    for (int i = 1; i < 16; ++i) mm = fmaxf(mm, red[r * 16 + i]);
    rowm[r] = mm;
  }
  __syncthreads();
  const float mm = rowm[r];
  float sum = 0.f;
  for (int j = 0; j < 256; ++j) sum += __expf(rp[j] - mm);
  red[r * 16 + sub] = sum;
  __syncthreads();
  if (sub == 0) {
    float ss = 0.f;
    for (int i = 0; i < 16; ++i) ss += red[r * 16 + i];
    rowinv[r] = 1.0f / ss;
  }
  __syncthreads();
  const float inv = rowinv[r];
  _Float16* Pr = P + ((size_t)b * NPIX + mb + r) * NPIX + sub * 256;
  for (int j = 0; j < 256; ++j)
    Pr[j] = (_Float16)(__expf(rp[j] - mm) * inv);
}

// ---------------- 5) O = P * V   (wave: 32 pixels x 64 channels) ------------
__global__ __launch_bounds__(128) void attn_out_gemm(const _Float16* __restrict__ P,
                                                     const _Float16* __restrict__ vT,
                                                     _Float16* __restrict__ O) {
  const int lane = threadIdx.x & 31, wave = threadIdx.x >> 5;
  const int nb = (blockIdx.y * 4 + wave) * 64;
  const int mb = blockIdx.x * 32;
  const int b  = blockIdx.z;
  const int lr = lane & 15;
  const _Float16* A = P  + (size_t)b * NPIX * NPIX;
  const _Float16* V = vT + (size_t)b * CC * NPIX;
  v8f acc0[4] = {}, acc1[4] = {};
  for (int kb2 = 0; kb2 < NPIX; kb2 += 32) {
    // prefetch next K-panel of the P rows (the unique data stream)
    __builtin_prefetch(A + (size_t)(mb + lr) * NPIX + kb2 + 32, 0, 1);
    __builtin_prefetch(A + (size_t)(mb + 16 + lr) * NPIX + kb2 + 32, 0, 1);
    v16h a0 = frag_f16(A, NPIX, mb,      kb2, lane);
    v16h a1 = frag_f16(A, NPIX, mb + 16, kb2, lane);
#pragma unroll
    for (int t = 0; t < 4; ++t) {
      v16h bf = frag_f16(V, NPIX, nb + t * 16, kb2, lane);
      acc0[t] = WMMA_F16(a0, bf, acc0[t]);
      acc1[t] = WMMA_F16(a1, bf, acc1[t]);
    }
  }
  const int hf = lane >> 4;
#pragma unroll
  for (int t = 0; t < 4; ++t) {
    const int n = nb + t * 16 + lr;
#pragma unroll
    for (int v = 0; v < 8; ++v) {
      const int m = mb + v + 8 * hf;
      O[((size_t)b * NPIX + m) * CC + n]      = (_Float16)acc0[t][v];
      O[((size_t)b * NPIX + m + 16) * CC + n] = (_Float16)acc1[t][v];
    }
  }
}

// ---------------- 6) out = x + Wo*O + bo ------------------------------------
__global__ __launch_bounds__(128) void proj_residual(const _Float16* __restrict__ O,
                                                     const _Float16* __restrict__ woh,
                                                     const float* __restrict__ bo,
                                                     const float* __restrict__ x,
                                                     float* __restrict__ out) {
  const int lane = threadIdx.x & 31, wave = threadIdx.x >> 5;
  const int nb = (blockIdx.y * 4 + wave) * 64;   // cout
  const int mb = blockIdx.x * 16;                // pixel
  const int b  = blockIdx.z;
  const _Float16* A = O + (size_t)b * NPIX * CC;
  v8f acc[4] = {};
  for (int kb2 = 0; kb2 < CC; kb2 += 32) {
    v16h a = frag_f16(A, CC, mb, kb2, lane);
#pragma unroll
    for (int t = 0; t < 4; ++t) {
      v16h bf = frag_f16(woh, CC, nb + t * 16, kb2, lane);
      acc[t] = WMMA_F16(a, bf, acc[t]);
    }
  }
  const int lr = lane & 15, hf = lane >> 4;
#pragma unroll
  for (int t = 0; t < 4; ++t) {
    const int n = nb + t * 16 + lr;
    const float bias = bo[n];
#pragma unroll
    for (int v = 0; v < 8; ++v) {
      const int m = mb + v + 8 * hf;
      const size_t idx = ((size_t)b * CC + n) * NPIX + m;
      out[idx] = x[idx] + acc[t][v] + bias;
    }
  }
}

// ---------------------------------------------------------------------------
extern "C" void kernel_launch(void* const* d_in, const int* in_sizes, int n_in,
                              void* d_out, int out_size, void* d_ws, size_t ws_size,
                              hipStream_t stream) {
  (void)in_sizes; (void)n_in; (void)out_size; (void)ws_size;
  const float* x    = (const float*)d_in[0];
  const float* gn_w = (const float*)d_in[1];
  const float* gn_b = (const float*)d_in[2];
  const float* wq   = (const float*)d_in[3];
  const float* bq   = (const float*)d_in[4];
  const float* wk   = (const float*)d_in[5];
  const float* bk   = (const float*)d_in[6];
  const float* wv   = (const float*)d_in[7];
  const float* bv   = (const float*)d_in[8];
  const float* wo   = (const float*)d_in[9];
  const float* bo   = (const float*)d_in[10];
  float* out = (float*)d_out;

  char* ws = (char*)d_ws;
  const size_t SZ16 = (size_t)BB * NPIX * CC * sizeof(_Float16);  // 16 MB
  const size_t SZP  = (size_t)BB * NPIX * NPIX * sizeof(_Float16);
  float*    stats = (float*)ws;
  _Float16* xn = (_Float16*)(ws + 1024);
  _Float16* q  = (_Float16*)(ws + 1024 + SZ16);
  _Float16* k  = (_Float16*)(ws + 1024 + 2 * SZ16);
  _Float16* vT = (_Float16*)(ws + 1024 + 3 * SZ16);
  _Float16* P  = (_Float16*)(ws + 1024 + 4 * SZ16);
  _Float16* O  = (_Float16*)(ws + 1024 + 4 * SZ16 + SZP);
  _Float16* wh = (_Float16*)(ws + 1024 + 5 * SZ16 + SZP);   // 4 x [C,C] f16

  prep_weights<<<(4 * CC * CC / 4) / 256, 256, 0, stream>>>(wq, wk, wv, wo, wh);
  gn_stats<<<BB * GRPS, 256, 0, stream>>>(x, stats);
  gn_apply<<<BB * CC, 256, 0, stream>>>(x, gn_w, gn_b, stats, xn);
  qkv_gemm<<<dim3(NPIX / 16, 6, BB), 128, 0, stream>>>(xn, wh, bq, bk, bv,
                                                       q, k, vT);
  attn_scores_softmax<<<dim3(NPIX / 16, BB), 256, 16 * NPIX * sizeof(float),
                        stream>>>(q, k, P);
  attn_out_gemm<<<dim3(NPIX / 32, 2, BB), 128, 0, stream>>>(P, vT, O);
  proj_residual<<<dim3(NPIX / 16, 2, BB), 128, 0, stream>>>(
      O, wh + 3 * (size_t)CC * CC, bo, x, out);
}
